// QTranBase_74783970557989
// MI455X (gfx1250) — compile-verified, matrix operand order checked by reading, not run on hardware
//
#include <hip/hip_runtime.h>

typedef __attribute__((ext_vector_type(16))) __bf16 v16bf;
typedef __attribute__((ext_vector_type(8)))  __bf16 v8bf;
typedef __attribute__((ext_vector_type(8)))  float  v8f;
typedef __attribute__((ext_vector_type(4)))  float  v4f;

#define N_ROWS   32768
#define D_EMBED  512
#define LDH      (512 + 8)   // padded LDS row stride (halves); 1040B = 16B-aligned, rotates banks by 4/row

// ---------------------------------------------------------------------------
// Pack weights: f32 W[K][512] row-major -> bf16 in WMMA B-fragment order.
// Fragment f = nt*(K/32)+kt holds 512 halves; lane L supplies column
// n = nt*16 + (L&15), K-offsets kb..kb+7 and kb+16..kb+23 with kb=(L>>4)*8.
// GEMM-side load is then one contiguous 32B read per lane.
// ---------------------------------------------------------------------------
__global__ __launch_bounds__(256) void pack_w_kernel(
    const float* __restrict__ W, __bf16* __restrict__ out, int K)
{
  int idx = blockIdx.x * 256 + threadIdx.x;
  if (idx >= K * D_EMBED) return;
  int f    = idx >> 9;          // fragment index
  int r    = idx & 511;
  int lane = r >> 4;
  int j    = r & 15;
  int ktiles = K >> 5;
  int kt = f % ktiles;
  int nt = f / ktiles;
  int kb = (lane >> 4) << 3;
  int koff = (j < 8) ? (kb + j) : (kb + 16 + (j - 8));
  int k = (kt << 5) + koff;
  int n = (nt << 4) + (lane & 15);
  out[idx] = (__bf16)W[(size_t)k * D_EMBED + n];
}

// ---------------------------------------------------------------------------
// Fused 3-layer MLP: out = relu(relu(A@W1+b1)@W2+b2)@w3 + b3
// Block = 256 threads (8 waves), 32 rows per block.
// Wave (wm = wave&1, wn = wave>>1): rows wm*16..+16, cols wn*128..+128
// (8 n-tiles -> 8 v8f accumulators per wave).
// ---------------------------------------------------------------------------
__global__ __launch_bounds__(256) void mlp3_fused_kernel(
    const float* __restrict__ A0, const float* __restrict__ A1, int kSplit, int K1,
    const __bf16* __restrict__ W1p, const float* __restrict__ b1,
    const __bf16* __restrict__ W2p, const float* __restrict__ b2,
    const float* __restrict__ w3, const float* __restrict__ b3,
    float* __restrict__ out)
{
  __shared__ __align__(16) __bf16 hbuf[32 * LDH];
  __shared__ float rowsum[32];

  const int lane = threadIdx.x & 31;
  const int wave = threadIdx.x >> 5;
  const int wm   = wave & 1;
  const int wn   = wave >> 1;
  const int r_l  = lane & 15;
  const int hh   = lane >> 4;       // half-wave select
  const int kb_l = hh << 3;         // per-lane K base inside a 32-K tile
  const int row0 = blockIdx.x * 32;

  if (threadIdx.x < 32) rowsum[threadIdx.x] = 0.f;

  v8f zero = {};
  v8f c[8];
  #pragma unroll
  for (int t = 0; t < 8; ++t) c[t] = zero;

  // ------------------------- layer 1 (K = K1) -----------------------------
  const int ktiles1 = K1 >> 5;
  const int myrow = row0 + wm * 16 + r_l;
  for (int kt = 0; kt < ktiles1; ++kt) {
    const int k0 = kt << 5;
    const float* Ab; int kc;
    if (k0 < kSplit) { Ab = A0; kc = k0; } else { Ab = A1; kc = k0 - kSplit; }
    const float* p = Ab + (size_t)myrow * 512 + kc + kb_l;
    v4f x0 = *(const v4f*)(p);
    v4f x1 = *(const v4f*)(p + 4);
    v4f x2 = *(const v4f*)(p + 16);
    v4f x3 = *(const v4f*)(p + 20);
    v16bf a;
    #pragma unroll
    for (int j = 0; j < 4; ++j) {
      a[j]      = (__bf16)x0[j];
      a[4 + j]  = (__bf16)x1[j];
      a[8 + j]  = (__bf16)x2[j];
      a[12 + j] = (__bf16)x3[j];
    }
    #pragma unroll
    for (int t = 0; t < 8; ++t) {
      const int nt = wn * 8 + t;
      v16bf b = *(const v16bf*)(W1p + ((size_t)nt * ktiles1 + kt) * 512 + lane * 16);
      c[t] = __builtin_amdgcn_wmma_f32_16x16x32_bf16(
                 false, a, false, b, (short)0, c[t], false, false);
    }
  }

  // bias + relu -> LDS (bf16), C layout: row = r + hh*8, col = r_l
  #pragma unroll
  for (int t = 0; t < 8; ++t) {
    const int col = (wn * 8 + t) * 16 + r_l;
    const float bv = b1[col];
    #pragma unroll
    for (int r = 0; r < 8; ++r) {
      const int rl = wm * 16 + hh * 8 + r;
      float v = c[t][r] + bv;
      v = v > 0.f ? v : 0.f;
      hbuf[rl * LDH + col] = (__bf16)v;
    }
  }
  __syncthreads();

  // ------------------------- layer 2 (K = 512) ----------------------------
  #pragma unroll
  for (int t = 0; t < 8; ++t) c[t] = zero;
  const int arow = wm * 16 + r_l;
  for (int kt = 0; kt < 16; ++kt) {
    const __bf16* p = &hbuf[arow * LDH + (kt << 5) + kb_l];
    v8bf lo = *(const v8bf*)(p);
    v8bf hi = *(const v8bf*)(p + 16);
    v16bf a;
    #pragma unroll
    for (int j = 0; j < 8; ++j) { a[j] = lo[j]; a[8 + j] = hi[j]; }
    #pragma unroll
    for (int t = 0; t < 8; ++t) {
      const int nt = wn * 8 + t;
      v16bf b = *(const v16bf*)(W2p + ((size_t)nt * 16 + kt) * 512 + lane * 16);
      c[t] = __builtin_amdgcn_wmma_f32_16x16x32_bf16(
                 false, a, false, b, (short)0, c[t], false, false);
    }
  }

  // --------------- fused layer 3: (relu(c+b2)) . w3  ----------------------
  float b2v[8], w3v[8];
  #pragma unroll
  for (int t = 0; t < 8; ++t) {
    const int col = (wn * 8 + t) * 16 + r_l;
    b2v[t] = b2[col];
    w3v[t] = w3[col];
  }
  #pragma unroll
  for (int r = 0; r < 8; ++r) {
    float acc = 0.f;
    #pragma unroll
    for (int t = 0; t < 8; ++t) {
      float v = c[t][r] + b2v[t];
      v = v > 0.f ? v : 0.f;
      acc += v * w3v[t];
    }
    atomicAdd(&rowsum[wm * 16 + hh * 8 + r], acc);   // ds_add_f32 cross-wave reduce
  }
  __syncthreads();
  if (threadIdx.x < 32)
    out[row0 + threadIdx.x] = rowsum[threadIdx.x] + b3[0];
}

// ---------------------------------------------------------------------------
extern "C" void kernel_launch(void* const* d_in, const int* in_sizes, int n_in,
                              void* d_out, int out_size, void* d_ws, size_t ws_size,
                              hipStream_t stream) {
  (void)in_sizes; (void)n_in; (void)out_size; (void)ws_size;
  const float* states  = (const float*)d_in[0];
  const float* actions = (const float*)d_in[1];
  const float* Qw1 = (const float*)d_in[2];
  const float* Qb1 = (const float*)d_in[3];
  const float* Qw2 = (const float*)d_in[4];
  const float* Qb2 = (const float*)d_in[5];
  const float* Qw3 = (const float*)d_in[6];
  const float* Qb3 = (const float*)d_in[7];
  const float* Vw1 = (const float*)d_in[8];
  const float* Vb1 = (const float*)d_in[9];
  const float* Vw2 = (const float*)d_in[10];
  const float* Vb2 = (const float*)d_in[11];
  const float* Vw3 = (const float*)d_in[12];
  const float* Vb3 = (const float*)d_in[13];
  float* out = (float*)d_out;

  char* ws = (char*)d_ws;
  __bf16* Qw1p = (__bf16*)ws;  ws += (size_t)1024 * 512 * 2;
  __bf16* Qw2p = (__bf16*)ws;  ws += (size_t)512 * 512 * 2;
  __bf16* Vw1p = (__bf16*)ws;  ws += (size_t)512 * 512 * 2;
  __bf16* Vw2p = (__bf16*)ws;  ws += (size_t)512 * 512 * 2;

  pack_w_kernel<<<(1024 * 512) / 256, 256, 0, stream>>>(Qw1, Qw1p, 1024);
  pack_w_kernel<<<(512  * 512) / 256, 256, 0, stream>>>(Qw2, Qw2p, 512);
  pack_w_kernel<<<(512  * 512) / 256, 256, 0, stream>>>(Vw1, Vw1p, 512);
  pack_w_kernel<<<(512  * 512) / 256, 256, 0, stream>>>(Vw2, Vw2p, 512);

  // Q network: A = concat(states, actions) handled by kSplit at K=512
  mlp3_fused_kernel<<<N_ROWS / 32, 256, 0, stream>>>(
      states, actions, 512, 1024, Qw1p, Qb1, Qw2p, Qb2, Qw3, Qb3, out);
  // V network: K1 = 512, A1 never reached (kSplit == K1)
  mlp3_fused_kernel<<<N_ROWS / 32, 256, 0, stream>>>(
      states, states, 512, 512, Vw1p, Vb1, Vw2p, Vb2, Vw3, Vb3, out + N_ROWS);
}
